// ConvAndLifMnist_80685255623163
// MI455X (gfx1250) — matrix operand order repaired and checked by reading, not compiled, not run on hardware
//
#include <hip/hip_runtime.h>
#include <hip/hip_bf16.h>
#include <stdint.h>

// ---------------------------------------------------------------------------
// SNN (conv->BN->relu->pool x2, rate-coded spikes, 10-step LIF MLP) for
// MI455X / gfx1250.  Per-step GEMMs run on v_wmma_f32_16x16x32_f16 with f32
// accumulators (membranes) resident in VGPRs across the 10-step scan.
// Spikes live as packed bits (2.6MB); B-matrices are pre-packed row-major
// [N=16][K=32] per tile so each lane's WMMA-B operand is two b128 loads.
// W2/W3 are staged to LDS via the Tensor Data Mover (tensor_load_to_lds +
// s_wait_tensorcnt), with an idempotent cooperative copy as a compile-only-
// environment safety net.  Workspace budget ~37.6 MB.
// ---------------------------------------------------------------------------

typedef __attribute__((ext_vector_type(16))) _Float16 v16h;
typedef __attribute__((ext_vector_type(8)))  float    v8f;
typedef unsigned int u32x4 __attribute__((ext_vector_type(4)));
typedef int          i32x4 __attribute__((ext_vector_type(4)));
typedef int          i32x8 __attribute__((ext_vector_type(8)));

union ABReg { v16h v; uint32_t u[8]; };

static __device__ inline v8f wmma_f16(const ABReg& a, const ABReg& b, v8f c) {
  // 8 args: (neg_a, A, neg_b, B, c_mod, C, reuse_a, reuse_b)
  return __builtin_amdgcn_wmma_f32_16x16x32_f16(false, a.v, false, b.v,
                                                (short)0, c, false, false);
}

// ---------------- sizes / workspace layout ----------------
#define BATCH 8192
#define NSTEP 10

#define OFF_STATS 0u            // 256 floats (1 KB)
#define OFF_X1   1024u                          // 8192*6*12*12 f32 = 28,311,552 B
#define OFF_FLAT 28312576u                      // 8192*256 f32     =  8,388,608 B
#define OFF_BITS 36701184u                      // 10*8192*8 u32    =  2,621,440 B
#define OFF_W1P  39322624u                      // 32768 f16 = 65,536 B
#define OFF_W2P  39388160u                      // 12288 f16 = 24,576 B  (contiguous
#define OFF_W3P  39412736u                      //  1536 f16 =  3,072 B   with W2P)
#define OFF_B1P  39415808u                      // 128 f32
#define OFF_B2P  39416320u                      //  96 f32
#define OFF_B3P  39416704u                      //  16 f32

#define W23_HALFS 13824                          // 12288 + 1536, one TDM transfer

// ---------------- threefry 2x32-20 (JAX RNG family, key = 42) -------------
static __device__ inline uint32_t rotl32(uint32_t x, int n) {
  return (x << n) | (x >> (32 - n));
}
static __device__ inline void threefry2x32(uint32_t c0, uint32_t c1,
                                           uint32_t& o0, uint32_t& o1) {
  const uint32_t k0 = 0u, k1 = 42u, k2 = 0x1BD11BDAu ^ k0 ^ k1;
  uint32_t x0 = c0 + k0, x1 = c1 + k1;
#define TF_RND(r) { x0 += x1; x1 = rotl32(x1, r); x1 ^= x0; }
  TF_RND(13) TF_RND(15) TF_RND(26) TF_RND(6)
  x0 += k1; x1 += k2 + 1u;
  TF_RND(17) TF_RND(29) TF_RND(16) TF_RND(24)
  x0 += k2; x1 += k0 + 2u;
  TF_RND(13) TF_RND(15) TF_RND(26) TF_RND(6)
  x0 += k0; x1 += k1 + 3u;
  TF_RND(17) TF_RND(29) TF_RND(16) TF_RND(24)
  x0 += k1; x1 += k2 + 4u;
  TF_RND(13) TF_RND(15) TF_RND(26) TF_RND(6)
  x0 += k2; x1 += k0 + 5u;
#undef TF_RND
  o0 = x0; o1 = x1;
}
static __device__ inline float bits_to_uniform(uint32_t b) {
  return __uint_as_float(0x3F800000u | (b >> 9)) - 1.0f;
}

// =====================  conv1 + BN stats  ==================================
__global__ __launch_bounds__(256) void k_conv1_stats(
    const float* __restrict__ img, const float* __restrict__ c1w,
    const float* __restrict__ c1b, float* __restrict__ stats) {
  int idx = blockIdx.x * 256 + threadIdx.x;     // < 8192*576
  int b = idx / 576, r = idx % 576;
  int oy = r / 24, ox = r % 24;
  const float* ip = img + b * 784 + oy * 28 + ox;
  float patch[25];
#pragma unroll
  for (int i = 0; i < 5; ++i)
#pragma unroll
    for (int j = 0; j < 5; ++j) patch[i * 5 + j] = ip[i * 28 + j];

  float sum[6], sq[6];
#pragma unroll
  for (int c = 0; c < 6; ++c) {
    float acc = c1b[c];
#pragma unroll
    for (int k = 0; k < 25; ++k) acc = fmaf(patch[k], c1w[c * 25 + k], acc);
    sum[c] = acc; sq[c] = acc * acc;
  }
  __shared__ float lacc[12];
  if (threadIdx.x < 12) lacc[threadIdx.x] = 0.0f;
  __syncthreads();
  int lane = threadIdx.x & 31;
#pragma unroll
  for (int c = 0; c < 6; ++c) {
    float s = sum[c], q = sq[c];
#pragma unroll
    for (int off = 16; off > 0; off >>= 1) {
      s += __shfl_xor(s, off);
      q += __shfl_xor(q, off);
    }
    if (lane == 0) { atomicAdd(&lacc[c], s); atomicAdd(&lacc[6 + c], q); }
  }
  __syncthreads();
  if (threadIdx.x < 12) {
    int di = (threadIdx.x < 6) ? threadIdx.x : (threadIdx.x + 2); // sq at 8..13
    atomicAdd(&stats[di], lacc[threadIdx.x]);
  }
}

__global__ __launch_bounds__(32) void k_finalize1(
    const float* __restrict__ g1, const float* __restrict__ bn1b,
    float* __restrict__ stats) {
  int c = threadIdx.x;
  if (c < 6) {
    const float N = 8192.0f * 576.0f;
    float mu = stats[c] / N;
    float var = stats[8 + c] / N - mu * mu;
    float sc = g1[c] * rsqrtf(var + 1e-5f);
    stats[48 + c] = sc;
    stats[56 + c] = bn1b[c] - mu * sc;
  }
}

// conv1 recompute + BN + relu + 2x2 maxpool -> x1 [B,6,12,12]
__global__ __launch_bounds__(256) void k_conv1_bn_pool(
    const float* __restrict__ img, const float* __restrict__ c1w,
    const float* __restrict__ c1b, const float* __restrict__ stats,
    float* __restrict__ x1) {
  int idx = blockIdx.x * 256 + threadIdx.x;     // < 8192*144
  int b = idx / 144, r = idx % 144;
  int py = r / 12, px = r % 12;
  const float* ip = img + b * 784 + (2 * py) * 28 + 2 * px;
  float patch[36];
#pragma unroll
  for (int i = 0; i < 6; ++i)
#pragma unroll
    for (int j = 0; j < 6; ++j) patch[i * 6 + j] = ip[i * 28 + j];
#pragma unroll
  for (int c = 0; c < 6; ++c) {
    float sc = stats[48 + c], sh = stats[56 + c], bias = c1b[c];
    float mx = -3.0e38f;
#pragma unroll
    for (int dy = 0; dy < 2; ++dy)
#pragma unroll
      for (int dx = 0; dx < 2; ++dx) {
        float acc = bias;
#pragma unroll
        for (int i = 0; i < 5; ++i)
#pragma unroll
          for (int j = 0; j < 5; ++j)
            acc = fmaf(patch[(dy + i) * 6 + dx + j], c1w[c * 25 + i * 5 + j], acc);
        mx = fmaxf(mx, fmaf(sc, acc, sh));
      }
    x1[(b * 6 + c) * 144 + py * 12 + px] = fmaxf(mx, 0.0f);
  }
}

// =====================  conv2 + BN stats  ==================================
__global__ __launch_bounds__(256) void k_conv2_stats(
    const float* __restrict__ x1, const float* __restrict__ c2w,
    const float* __restrict__ c2b, float* __restrict__ stats) {
  __shared__ float sw[2416];
  for (int i = threadIdx.x; i < 2400; i += 256) sw[i] = c2w[i];
  if (threadIdx.x < 16) sw[2400 + threadIdx.x] = c2b[threadIdx.x];
  __syncthreads();

  int idx = blockIdx.x * 256 + threadIdx.x;     // < 8192*64
  int b = idx / 64, r = idx % 64;
  int oy = r / 8, ox = r % 8;

  float acc[16];
#pragma unroll
  for (int o = 0; o < 16; ++o) acc[o] = sw[2400 + o];
  for (int c = 0; c < 6; ++c) {
    float p25[25];
    const float* ip = x1 + (b * 6 + c) * 144 + oy * 12 + ox;
#pragma unroll
    for (int i = 0; i < 5; ++i)
#pragma unroll
      for (int j = 0; j < 5; ++j) p25[i * 5 + j] = ip[i * 12 + j];
#pragma unroll
    for (int o = 0; o < 16; ++o)
#pragma unroll
      for (int k = 0; k < 25; ++k)
        acc[o] = fmaf(p25[k], sw[o * 150 + c * 25 + k], acc[o]);
  }
  __shared__ float lacc[32];
  if (threadIdx.x < 32) lacc[threadIdx.x] = 0.0f;
  __syncthreads();
  int lane = threadIdx.x & 31;
#pragma unroll
  for (int o = 0; o < 16; ++o) {
    float s = acc[o], q = acc[o] * acc[o];
#pragma unroll
    for (int off = 16; off > 0; off >>= 1) {
      s += __shfl_xor(s, off);
      q += __shfl_xor(q, off);
    }
    if (lane == 0) { atomicAdd(&lacc[o], s); atomicAdd(&lacc[16 + o], q); }
  }
  __syncthreads();
  if (threadIdx.x < 32) atomicAdd(&stats[16 + threadIdx.x], lacc[threadIdx.x]);
}

__global__ __launch_bounds__(32) void k_finalize2(
    const float* __restrict__ g2, const float* __restrict__ bn2b,
    float* __restrict__ stats) {
  int c = threadIdx.x;
  if (c < 16) {
    const float N = 8192.0f * 64.0f;
    float mu = stats[16 + c] / N;
    float var = stats[32 + c] / N - mu * mu;
    float sc = g2[c] * rsqrtf(var + 1e-5f);
    stats[64 + c] = sc;
    stats[80 + c] = bn2b[c] - mu * sc;
  }
}

// conv2 recompute + BN + relu + pool -> flat [B,256] (idx = c*16+py*4+px)
__global__ __launch_bounds__(256) void k_conv2_bn_pool(
    const float* __restrict__ x1, const float* __restrict__ c2w,
    const float* __restrict__ c2b, const float* __restrict__ stats,
    float* __restrict__ flat) {
  __shared__ float sw[2416];
  for (int i = threadIdx.x; i < 2400; i += 256) sw[i] = c2w[i];
  if (threadIdx.x < 16) sw[2400 + threadIdx.x] = c2b[threadIdx.x];
  __syncthreads();

  int idx = blockIdx.x * 256 + threadIdx.x;     // < 8192*16
  int b = idx / 16, r = idx % 16;
  int py = r / 4, px = r % 4;

  float acc[64];                                // [o][dy*2+dx]
#pragma unroll
  for (int o = 0; o < 16; ++o)
#pragma unroll
    for (int q = 0; q < 4; ++q) acc[o * 4 + q] = sw[2400 + o];

  for (int c = 0; c < 6; ++c) {
    float p36[36];
    const float* ip = x1 + (b * 6 + c) * 144 + (2 * py) * 12 + 2 * px;
#pragma unroll
    for (int i = 0; i < 6; ++i)
#pragma unroll
      for (int j = 0; j < 6; ++j) p36[i * 6 + j] = ip[i * 12 + j];
#pragma unroll
    for (int o = 0; o < 16; ++o)
#pragma unroll
      for (int dy = 0; dy < 2; ++dy)
#pragma unroll
        for (int dx = 0; dx < 2; ++dx) {
          float a = acc[o * 4 + dy * 2 + dx];
#pragma unroll
          for (int i = 0; i < 5; ++i)
#pragma unroll
            for (int j = 0; j < 5; ++j)
              a = fmaf(p36[(dy + i) * 6 + dx + j], sw[o * 150 + c * 25 + i * 5 + j], a);
          acc[o * 4 + dy * 2 + dx] = a;
        }
  }
#pragma unroll
  for (int o = 0; o < 16; ++o) {
    float sc = stats[64 + o], sh = stats[80 + o];
    float mx = -3.0e38f;
#pragma unroll
    for (int q = 0; q < 4; ++q) mx = fmaxf(mx, fmaf(sc, acc[o * 4 + q], sh));
    flat[b * 256 + o * 16 + py * 4 + px] = fmaxf(mx, 0.0f);
  }
}

// =====================  spike bits (rate coding)  ==========================
__global__ __launch_bounds__(256) void k_spikegen(
    const float* __restrict__ flat, uint32_t* __restrict__ bits) {
  int idx = blockIdx.x * 256 + threadIdx.x;     // < 10*8192*8
  int kd = idx & 7;
  int b = (idx >> 3) & 8191;
  const float* pf = flat + b * 256 + kd * 32;
  uint64_t ebase = (uint64_t)idx * 32u;
  uint32_t mask = 0u;
#pragma unroll
  for (int p = 0; p < 16; ++p) {
    uint64_t e = ebase + 2u * (uint32_t)p;
    uint32_t r0, r1;
    threefry2x32((uint32_t)e, (uint32_t)(e >> 32), r0, r1);
    float u0 = bits_to_uniform(r0), u1 = bits_to_uniform(r1);
    float p0 = fminf(fmaxf(pf[2 * p], 0.0f), 1.0f);
    float p1 = fminf(fmaxf(pf[2 * p + 1], 0.0f), 1.0f);
    if (u0 < p0) mask |= 1u << (2 * p);
    if (u1 < p1) mask |= 1u << (2 * p + 1);
  }
  bits[idx] = mask;
}

// =====================  weight packing into WMMA B order ===================
// B tile (K=32 x N=16) stored column-major: half index = dn*32 + dk.
// WMMA-B register layout (lane l holds N=l%16, K=8h+s / 16+8h+s-8) is then
// two b128 loads per lane.
__global__ __launch_bounds__(256) void k_pack_weights(
    const float* __restrict__ w1, const float* __restrict__ b1,
    const float* __restrict__ w2, const float* __restrict__ b2,
    const float* __restrict__ w3, const float* __restrict__ b3,
    _Float16* __restrict__ W1p, _Float16* __restrict__ W2p,
    _Float16* __restrict__ W3p, float* __restrict__ b1p,
    float* __restrict__ b2p, float* __restrict__ b3p) {
  int idx = blockIdx.x * 256 + threadIdx.x;
  if (idx < 32768) {                                   // W1: K=256, N=128
    int tile = idx >> 9, w = idx & 511;
    int kt = tile >> 3, nt = tile & 7;
    int dk = w & 31, dn = w >> 5;
    int k = kt * 32 + dk, n = nt * 16 + dn;
    float v = (n < 120) ? w1[n * 256 + k] : 0.0f;
    W1p[idx] = (_Float16)v;
  } else if (idx < 45056) {                            // W2: K=128, N=96
    int i = idx - 32768;
    int tile = i >> 9, w = i & 511;
    int kt = tile / 6, nt = tile % 6;
    int dk = w & 31, dn = w >> 5;
    int k = kt * 32 + dk, n = nt * 16 + dn;
    float v = (k < 120 && n < 84) ? w2[n * 120 + k] : 0.0f;
    W2p[i] = (_Float16)v;
  } else if (idx < 46592) {                            // W3: K=96, N=16
    int i = idx - 45056;
    int kt = i >> 9, w = i & 511;
    int dk = w & 31, dn = w >> 5;
    int k = kt * 32 + dk, n = dn;
    float v = (k < 84 && n < 10) ? w3[n * 84 + k] : 0.0f;
    W3p[i] = (_Float16)v;
  } else if (idx < 46720) {
    int i = idx - 46592; b1p[i] = (i < 120) ? b1[i] : 0.0f;
  } else if (idx < 46816) {
    int i = idx - 46720; b2p[i] = (i < 84) ? b2[i] : 0.0f;
  } else if (idx < 46832) {
    int i = idx - 46816; b3p[i] = (i < 10) ? b3[i] : 0.0f;
  }
}

// =====================  LIF scan with WMMA  ================================
static __device__ inline ABReg build_a_from_bits(uint32_t bw, int g) {
  ABReg a;
#pragma unroll
  for (int r = 0; r < 8; ++r) {
    int k0 = ((r < 4) ? 2 * r : 16 + 2 * (r - 4)) + 8 * g;
    uint32_t lo = (bw >> k0) & 1u;
    uint32_t hi = (bw >> (k0 + 1)) & 1u;
    a.u[r] = (lo ? 0x00003C00u : 0u) | (hi ? 0x3C000000u : 0u);
  }
  return a;
}
static __device__ inline ABReg load_a_lds(const _Float16* st, int stride,
                                          int row, int kBase, int g) {
  ABReg a;
  const uint4* p0 = (const uint4*)(st + row * stride + kBase + 8 * g);
  const uint4* p1 = (const uint4*)(st + row * stride + kBase + 16 + 8 * g);
  uint4 q0 = *p0, q1 = *p1;
  a.u[0] = q0.x; a.u[1] = q0.y; a.u[2] = q0.z; a.u[3] = q0.w;
  a.u[4] = q1.x; a.u[5] = q1.y; a.u[6] = q1.z; a.u[7] = q1.w;
  return a;
}
static __device__ inline ABReg load_b(const _Float16* tile, int lane) {
  ABReg b;
  const uint4* p = (const uint4*)(tile + (lane & 15) * 32 + 8 * (lane >> 4));
  uint4 q0 = p[0], q1 = p[2];
  b.u[0] = q0.x; b.u[1] = q0.y; b.u[2] = q0.z; b.u[3] = q0.w;
  b.u[4] = q1.x; b.u[5] = q1.y; b.u[6] = q1.z; b.u[7] = q1.w;
  return b;
}

// Issue one 1-D TDM transfer: W23_HALFS f16 from `gsrc` into LDS at `lds_addr`.
// D# field packing per CDNA5 ISA §8.3-8.6 (count=1, type=2 "image",
// data_size=2B, 1-D: tile_dim0=tensor_dim0=stride0=N, everything else 0).
static __device__ inline void tdm_load_w23(const _Float16* gsrc,
                                           uint32_t lds_addr) {
  uint64_t ga = (uint64_t)(uintptr_t)gsrc;
  u32x4 g0 = {0u, 0u, 0u, 0u};
  g0[0] = 1u;                                    // count=1, user descriptor
  g0[1] = lds_addr;                              // lds_addr (bytes)
  g0[2] = (uint32_t)ga;                          // global_addr[31:0]
  g0[3] = ((uint32_t)(ga >> 32) & 0x01FFFFFFu)   // global_addr[56:32]
          | (2u << 30);                          // type=2
  i32x8 g1 = {0, 0, 0, 0, 0, 0, 0, 0};
  g1[0] = (int)(1u << 16);                       // data_size=1 -> 2 bytes
  g1[1] = (int)((W23_HALFS & 0xFFFFu) << 16);    // tensor_dim0[15:0]
  g1[2] = (int)((W23_HALFS >> 16) | (1u << 16)); // tensor_dim0[31:16], dim1=1
  g1[3] = (int)((uint32_t)W23_HALFS << 16);      // tile_dim0
  g1[5] = (int)W23_HALFS;                        // tensor_dim0_stride[31:0]
  i32x4 gz = {0, 0, 0, 0};
#if __clang_major__ <= 22
  __builtin_amdgcn_tensor_load_to_lds(g0, g1, gz, gz, 0);
#else
  i32x8 gz8 = {0, 0, 0, 0, 0, 0, 0, 0};
  __builtin_amdgcn_tensor_load_to_lds(g0, g1, gz, gz, gz8, 0);
#endif
  __builtin_amdgcn_s_wait_tensorcnt(0);
}

__global__ __launch_bounds__(128) void k_lif_wmma(
    const uint32_t* __restrict__ bits, const _Float16* __restrict__ W1p,
    const _Float16* __restrict__ W2p, const _Float16* __restrict__ W3p,
    const float* __restrict__ b1p, const float* __restrict__ b2p,
    const float* __restrict__ b3p, float* __restrict__ out) {
  __shared__ __align__(16) _Float16 sW23[W23_HALFS];   // W2 tiles then W3 tiles
  __shared__ __align__(16) _Float16 stage[4][2048];

  const int tid = threadIdx.x;
  const int wave = tid >> 5, lane = tid & 31;

  // --- async staging of packed W2+W3 (contiguous in workspace) via TDM ---
  if (wave == 0) {
    tdm_load_w23(W2p, (uint32_t)(uintptr_t)&sW23[0]);
  }
  __syncthreads();
  {  // idempotent cooperative copy: descriptor semantics are not verifiable
     // in this compile-only environment, so re-write the same bytes.
    const uint32_t* gsrc = (const uint32_t*)W2p;
    uint32_t* ldst = (uint32_t*)sW23;
    for (int i = tid; i < W23_HALFS / 2; i += 128) ldst[i] = gsrc[i];
  }
  __syncthreads();

  const int row = lane & 15, g = lane >> 4;     // A: M=row ; C/D: N=row
  const int R = blockIdx.x * 64 + wave * 16;
  _Float16* st = &stage[wave][0];
  const _Float16* sW2 = &sW23[0];
  const _Float16* sW3 = &sW23[12288];

  float bias1v[8], bias2v[6];
#pragma unroll
  for (int nt = 0; nt < 8; ++nt) bias1v[nt] = b1p[nt * 16 + row];
#pragma unroll
  for (int nt = 0; nt < 6; ++nt) bias2v[nt] = b2p[nt * 16 + row];
  const float bias3v = b3p[row];

  v8f mem1[8], mem2[6], outA;
  const v8f vzero = {0.f, 0.f, 0.f, 0.f, 0.f, 0.f, 0.f, 0.f};
#pragma unroll
  for (int nt = 0; nt < 8; ++nt) mem1[nt] = vzero;
#pragma unroll
  for (int nt = 0; nt < 6; ++nt) mem2[nt] = vzero;
  outA = vzero;

  for (int t = 0; t < NSTEP; ++t) {
    if (t + 1 < NSTEP)   // pull next step's spike bits toward the WGP
      __builtin_prefetch(&bits[((t + 1) * BATCH + (R + row)) * 8], 0, 1);

    // ---------- fc1: mem1 = 0.9*mem1 - (mem1>1) + b1 + s_t @ W1 ----------
#pragma unroll
    for (int nt = 0; nt < 8; ++nt)
#pragma unroll
      for (int j = 0; j < 8; ++j) {
        float m = mem1[nt][j];
        mem1[nt][j] = 0.9f * m - ((m > 1.0f) ? 1.0f : 0.0f) + bias1v[nt];
      }
#pragma unroll
    for (int kt = 0; kt < 8; ++kt) {
      uint32_t bw = bits[(t * BATCH + (R + row)) * 8 + kt];
      ABReg a = build_a_from_bits(bw, g);
#pragma unroll
      for (int nt = 0; nt < 8; ++nt) {
        ABReg b = load_b(W1p + (kt * 8 + nt) * 512, lane);
        mem1[nt] = wmma_f16(a, b, mem1[nt]);
      }
    }
    // stage s1 = (mem1 > 1) as f16 rows [16 x 128]
#pragma unroll
    for (int nt = 0; nt < 8; ++nt)
#pragma unroll
      for (int jv = 0; jv < 8; ++jv)
        st[(jv + 8 * g) * 128 + nt * 16 + row] =
            (mem1[nt][jv] > 1.0f) ? (_Float16)1.0f : (_Float16)0.0f;

    // ---------- fc2: mem2 = 0.9*mem2 - (mem2>1) + b2 + s1 @ W2 ----------
#pragma unroll
    for (int nt = 0; nt < 6; ++nt)
#pragma unroll
      for (int j = 0; j < 8; ++j) {
        float m = mem2[nt][j];
        mem2[nt][j] = 0.9f * m - ((m > 1.0f) ? 1.0f : 0.0f) + bias2v[nt];
      }
#pragma unroll
    for (int kt = 0; kt < 4; ++kt) {
      ABReg a = load_a_lds(st, 128, row, kt * 32, g);
#pragma unroll
      for (int nt = 0; nt < 6; ++nt) {
        ABReg b = load_b(sW2 + (kt * 6 + nt) * 512, lane);
        mem2[nt] = wmma_f16(a, b, mem2[nt]);
      }
    }
    // stage s2 = (mem2 > 1) as f16 rows [16 x 96]
#pragma unroll
    for (int nt = 0; nt < 6; ++nt)
#pragma unroll
      for (int jv = 0; jv < 8; ++jv)
        st[(jv + 8 * g) * 96 + nt * 16 + row] =
            (mem2[nt][jv] > 1.0f) ? (_Float16)1.0f : (_Float16)0.0f;

    // ---------- fc3: outA += s2 @ W3 (bias added once at the end) --------
#pragma unroll
    for (int kt = 0; kt < 3; ++kt) {
      ABReg a = load_a_lds(st, 96, row, kt * 32, g);
      ABReg b = load_b(sW3 + kt * 512, lane);
      outA = wmma_f16(a, b, outA);
    }
  }

#pragma unroll
  for (int jv = 0; jv < 8; ++jv) {
    float v = outA[jv] + 10.0f * bias3v;
    if (row < 10) out[(R + jv + 8 * g) * 10 + row] = v;
  }
}

// =====================  host launcher  =====================================
extern "C" void kernel_launch(void* const* d_in, const int* in_sizes, int n_in,
                              void* d_out, int out_size, void* d_ws, size_t ws_size,
                              hipStream_t stream) {
  const float* images = (const float*)d_in[0];
  const float* c1w = (const float*)d_in[1];
  const float* c1b = (const float*)d_in[2];
  const float* g1  = (const float*)d_in[3];
  const float* bn1b = (const float*)d_in[4];
  const float* c2w = (const float*)d_in[5];
  const float* c2b = (const float*)d_in[6];
  const float* g2  = (const float*)d_in[7];
  const float* bn2b = (const float*)d_in[8];
  const float* w1 = (const float*)d_in[9];
  const float* b1 = (const float*)d_in[10];
  const float* w2 = (const float*)d_in[11];
  const float* b2 = (const float*)d_in[12];
  const float* w3 = (const float*)d_in[13];
  const float* b3 = (const float*)d_in[14];

  char* ws = (char*)d_ws;
  float*     stats = (float*)(ws + OFF_STATS);
  float*     x1    = (float*)(ws + OFF_X1);
  float*     flat  = (float*)(ws + OFF_FLAT);
  uint32_t*  bits  = (uint32_t*)(ws + OFF_BITS);
  _Float16*  W1p   = (_Float16*)(ws + OFF_W1P);
  _Float16*  W2p   = (_Float16*)(ws + OFF_W2P);
  _Float16*  W3p   = (_Float16*)(ws + OFF_W3P);
  float*     b1p   = (float*)(ws + OFF_B1P);
  float*     b2p   = (float*)(ws + OFF_B2P);
  float*     b3p   = (float*)(ws + OFF_B3P);

  hipMemsetAsync(stats, 0, 1024, stream);

  k_pack_weights<<<183, 256, 0, stream>>>(w1, b1, w2, b2, w3, b3,
                                          W1p, W2p, W3p, b1p, b2p, b3p);

  k_conv1_stats<<<18432, 256, 0, stream>>>(images, c1w, c1b, stats);
  k_finalize1<<<1, 32, 0, stream>>>(g1, bn1b, stats);
  k_conv1_bn_pool<<<4608, 256, 0, stream>>>(images, c1w, c1b, stats, x1);

  k_conv2_stats<<<2048, 256, 0, stream>>>(x1, c2w, c2b, stats);
  k_finalize2<<<1, 32, 0, stream>>>(g2, bn2b, stats);
  k_conv2_bn_pool<<<512, 256, 0, stream>>>(x1, c2w, c2b, stats, flat);

  k_spikegen<<<2560, 256, 0, stream>>>(flat, bits);

  k_lif_wmma<<<128, 128, 0, stream>>>(bits, W1p, W2p, W3p, b1p, b2p, b3p,
                                      (float*)d_out);
}
